// BottleneckAttention_73315091742837
// MI455X (gfx1250) — matrix operand order repaired and checked by reading, not compiled, not run on hardware
//
#include <hip/hip_runtime.h>

typedef __bf16 bf16_t;
typedef __attribute__((ext_vector_type(16))) __bf16 v16bf;
typedef __attribute__((ext_vector_type(8)))  __bf16 v8bf;
typedef __attribute__((ext_vector_type(4)))  __bf16 v4bf;
typedef __attribute__((ext_vector_type(8)))  float  v8f;

#define NN 4096
#define CCH 256

#define CAT16(lo, hi) __builtin_shufflevector((lo), (hi), 0,1,2,3,4,5,6,7,8,9,10,11,12,13,14,15)

__device__ __forceinline__ v8f wmma_bf16(v16bf a, v16bf b, v8f c) {
  return __builtin_amdgcn_wmma_f32_16x16x32_bf16(false, a, false, b, (short)0, c, false, false);
}

__device__ __forceinline__ v16bf load_ab(const bf16_t* lo, const bf16_t* hi) {
  return CAT16(*(const v8bf*)lo, *(const v8bf*)hi);
}

__device__ __forceinline__ void lds_fence() {
  asm volatile("s_wait_dscnt 0" ::: "memory");
}
__device__ __forceinline__ void compiler_fence() {
  asm volatile("" ::: "memory");
}

// ---------------------------------------------------------------------------
// K0: convert weights f32 -> bf16 row-major. gamma is folded into the LEFT
//     half of Wf (the half that multiplies the attention output), so the
//     fuse kernel needs no gamma and ob can be stored bf16.
// ---------------------------------------------------------------------------
__global__ void prep_kernel(const float* __restrict__ wq, const float* __restrict__ wk,
                            const float* __restrict__ wv, const float* __restrict__ wf,
                            const float* __restrict__ gptr,
                            bf16_t* __restrict__ wqb, bf16_t* __restrict__ wkb,
                            bf16_t* __restrict__ wvb, bf16_t* __restrict__ wfb) {
  int i = blockIdx.x * blockDim.x + threadIdx.x;
  if (i < 8192)   wqb[i] = (bf16_t)wq[i];
  if (i < 8192)   wkb[i] = (bf16_t)wk[i];
  if (i < 65536)  wvb[i] = (bf16_t)wv[i];
  if (i < 131072) {
    float g = gptr[0];
    float s = ((i & 511) < 256) ? g : 1.0f;
    wfb[i] = (bf16_t)(wf[i] * s);
  }
}

// ---------------------------------------------------------------------------
// K1: q,k projections, stored TRANSPOSED as [B][N][32] bf16.
//     q gets (val + bq) * (1/sqrt(32)) folded in; k gets (val + bk).
//     GEMM: out_t[n][o] = sum_c x[c,n] * W[o,c]
//     A[m=n][k=c] = x^T (gathered via LDS stage), B[k=c][col=o] = W rows.
// ---------------------------------------------------------------------------
__global__ __launch_bounds__(256) void projqk_kernel(
    const float* __restrict__ x, const bf16_t* __restrict__ wqb,
    const bf16_t* __restrict__ wkb, const float* __restrict__ bq,
    const float* __restrict__ bk, bf16_t* __restrict__ qt, bf16_t* __restrict__ kt) {
  __shared__ alignas(16) float xs[8][32 * 16];
  int wid = threadIdx.x >> 5, lane = threadIdx.x & 31;
  int h = lane >> 4, nl = lane & 15;
  int wg = blockIdx.x * 8 + wid;
  int b = wg >> 8, nt = wg & 255, n0 = nt << 4;
  float* xw = &xs[wid][0];

  v8f aq0, aq1, ak0, ak1;
#pragma unroll
  for (int e = 0; e < 8; ++e) { aq0[e] = 0.f; aq1[e] = 0.f; ak0[e] = 0.f; ak1[e] = 0.f; }

  for (int cb = 0; cb < CCH; cb += 32) {
    // stage x[cb..cb+32) x [n0..n0+16) into LDS (rows of 16 f32)
#pragma unroll
    for (int t = 0; t < 4; ++t) {
      int slot = t * 32 + lane;
      int c = slot >> 2, q4 = slot & 3;
      float4 vv = *(const float4*)(x + ((size_t)(b * CCH + cb + c)) * NN + n0 + q4 * 4);
      *(float4*)(xw + c * 16 + q4 * 4) = vv;
    }
    lds_fence();
    // A fragment: lane row n = n0+nl, K(=c) = 16h+idx contiguous, same order as B
    v16bf xa;
#pragma unroll
    for (int idx = 0; idx < 16; ++idx) {
      int K = 16 * h + idx;
      xa[idx] = (bf16_t)xw[K * 16 + nl];
    }
    compiler_fence();
    const bf16_t* wq0 = wqb + (size_t)nl * CCH + cb + 16 * h;
    const bf16_t* wq1 = wqb + (size_t)(nl + 16) * CCH + cb + 16 * h;
    const bf16_t* wk0 = wkb + (size_t)nl * CCH + cb + 16 * h;
    const bf16_t* wk1 = wkb + (size_t)(nl + 16) * CCH + cb + 16 * h;
    aq0 = wmma_bf16(xa, load_ab(wq0, wq0 + 8), aq0);
    aq1 = wmma_bf16(xa, load_ab(wq1, wq1 + 8), aq1);
    ak0 = wmma_bf16(xa, load_ab(wk0, wk0 + 8), ak0);
    ak1 = wmma_bf16(xa, load_ab(wk1, wk1 + 8), ak1);
  }
  float bq0 = bq[nl], bq1 = bq[nl + 16];
  float bk0v = bk[nl], bk1v = bk[nl + 16];
  const float qs = 0.17677669529663689f;  // 1/sqrt(32)
#pragma unroll
  for (int r = 0; r < 8; ++r) {
    int n = n0 + r + 8 * h;
    bf16_t* qrow = qt + ((size_t)(b * NN) + n) * 32;
    bf16_t* krow = kt + ((size_t)(b * NN) + n) * 32;
    qrow[nl]      = (bf16_t)((aq0[r] + bq0) * qs);
    qrow[16 + nl] = (bf16_t)((aq1[r] + bq1) * qs);
    krow[nl]      = (bf16_t)(ak0[r] + bk0v);
    krow[16 + nl] = (bf16_t)(ak1[r] + bk1v);
  }
}

// ---------------------------------------------------------------------------
// K2: v projection, stored [B][256][N] bf16.
//     D[c',n] = sum_c Wv[c',c] * x[c,n] + bv. A = Wv rows, B = x columns.
// ---------------------------------------------------------------------------
__global__ __launch_bounds__(256) void projv_kernel(
    const float* __restrict__ x, const bf16_t* __restrict__ wvb,
    const float* __restrict__ bv, bf16_t* __restrict__ vb) {
  __shared__ alignas(16) float xs[8][32 * 16];
  int wid = threadIdx.x >> 5, lane = threadIdx.x & 31;
  int h = lane >> 4, nl = lane & 15;
  int wg = blockIdx.x * 8 + wid;
  int b = wg >> 12;
  int ct = (wg >> 8) & 15;
  int nt = wg & 255, n0 = nt << 4;
  float* xw = &xs[wid][0];

  v8f acc;
#pragma unroll
  for (int e = 0; e < 8; ++e) acc[e] = 0.f;

  for (int cb = 0; cb < CCH; cb += 32) {
#pragma unroll
    for (int t = 0; t < 4; ++t) {
      int slot = t * 32 + lane;
      int c = slot >> 2, q4 = slot & 3;
      float4 vv = *(const float4*)(x + ((size_t)(b * CCH + cb + c)) * NN + n0 + q4 * 4);
      *(float4*)(xw + c * 16 + q4 * 4) = vv;
    }
    lds_fence();
    // B fragment: lane column n = n0+nl, K(=c) in table order 8h+s / 16+8h+(s-8), as A
    v16bf xb;
#pragma unroll
    for (int idx = 0; idx < 16; ++idx) {
      int K = ((idx < 8) ? 0 : 16) + 8 * h + (idx & 7);
      xb[idx] = (bf16_t)xw[K * 16 + nl];
    }
    compiler_fence();
    const bf16_t* wr = wvb + ((size_t)(ct * 16 + nl)) * CCH + cb;
    v16bf wa = load_ab(wr + 8 * h, wr + 16 + 8 * h);
    acc = wmma_bf16(wa, xb, acc);
  }
#pragma unroll
  for (int r = 0; r < 8; ++r) {
    int c = ct * 16 + r + 8 * h;
    vb[((size_t)b * CCH + c) * NN + n0 + nl] = (bf16_t)(acc[r] + bv[c]);
  }
}

// ---------------------------------------------------------------------------
// K3: fused flash-attention. One wave owns one (batch, 16-row i-tile),
//     keeps out[256 x 16] in 128 f32 accumulator VGPRs, streams j in
//     chunks of 32 with online softmax. The 128-VALU accumulator rescale is
//     skipped via a wave-uniform vote when the running max did not change
//     (the common case). Output written bf16 (gamma lives in Wf).
// ---------------------------------------------------------------------------
__global__ __launch_bounds__(256) void attn_kernel(
    const bf16_t* __restrict__ qt, const bf16_t* __restrict__ kt,
    const bf16_t* __restrict__ vb, bf16_t* __restrict__ ob) {
  __shared__ alignas(16) bf16_t pb[8][16 * 32];
  __shared__ float st[8][16];
  int wid = threadIdx.x >> 5, lane = threadIdx.x & 31;
  int h = lane >> 4, nl = lane & 15;
  int wg = blockIdx.x * 8 + wid;
  int b = wg >> 8, it = wg & 255;
  int i0 = it << 4;

  bf16_t* pw = &pb[wid][0];
  float* sw = &st[wid][0];

  // q A-fragment (row i = i0+nl, d runs 16h..16h+15 contiguous); 1/sqrt(d_k) pre-folded
  const bf16_t* qr = qt + ((size_t)(b * NN) + i0 + nl) * 32;
  v16bf qa = load_ab(qr + 16 * h, qr + 16 * h + 8);

  v8f acc[16];
#pragma unroll
  for (int ct = 0; ct < 16; ++ct)
#pragma unroll
    for (int e = 0; e < 8; ++e) acc[ct][e] = 0.f;

  float M[8], L[8];
#pragma unroll
  for (int r = 0; r < 8; ++r) { M[r] = -__builtin_inff(); L[r] = 0.f; }
  v8f zero;
#pragma unroll
  for (int e = 0; e < 8; ++e) zero[e] = 0.f;

  for (int jj = 0; jj < NN; jj += 32) {
    // two 16x16 score tiles: D[i][j], K = d = 32 in a single WMMA each
    v8f s0, s1;
    {
      const bf16_t* kr = kt + ((size_t)(b * NN) + jj + nl) * 32 + 16 * h;
      s0 = wmma_bf16(qa, load_ab(kr, kr + 8), zero);
    }
    {
      const bf16_t* kr = kt + ((size_t)(b * NN) + jj + 16 + nl) * 32 + 16 * h;
      s1 = wmma_bf16(qa, load_ab(kr, kr + 8), zero);
    }
    // online softmax per row i = r + 8h (16-lane halves)
    float alpha[8];
    bool grew = false;
#pragma unroll
    for (int r = 0; r < 8; ++r) {
      float t = fmaxf(s0[r], s1[r]);
      t = fmaxf(t, __shfl_xor(t, 1));
      t = fmaxf(t, __shfl_xor(t, 2));
      t = fmaxf(t, __shfl_xor(t, 4));
      t = fmaxf(t, __shfl_xor(t, 8));
      float mo = M[r];
      float mn = fmaxf(mo, t);
      alpha[r] = __expf(mo - mn);
      grew |= (mn > mo);
      M[r] = mn;
      float p0 = __expf(s0[r] - mn);
      float p1 = __expf(s1[r] - mn);
      s0[r] = p0; s1[r] = p1;
      float rs = p0 + p1;
      rs += __shfl_xor(rs, 1);
      rs += __shfl_xor(rs, 2);
      rs += __shfl_xor(rs, 4);
      rs += __shfl_xor(rs, 8);
      L[r] = L[r] * alpha[r] + rs;
    }
    bool resc = __any((int)grew);  // wave-uniform branch condition
    // transpose P into LDS rows [i][j32]
#pragma unroll
    for (int r = 0; r < 8; ++r) {
      int i = r + 8 * h;
      pw[i * 32 + nl]      = (bf16_t)s0[r];
      pw[i * 32 + 16 + nl] = (bf16_t)s1[r];
    }
    if (resc) {
#pragma unroll
      for (int r = 0; r < 8; ++r) sw[r + 8 * h] = alpha[r];  // broadcast by row
    }
    lds_fence();
    const bf16_t* pr = pw + nl * 32 + 16 * h;
    v16bf pf = load_ab(pr, pr + 8);  // B[k=j][col=i]
    if (resc) {
      float av = sw[nl];  // alpha for column i = nl (acc D-layout column)
#pragma unroll
      for (int ct = 0; ct < 16; ++ct)
#pragma unroll
        for (int e = 0; e < 8; ++e) acc[ct][e] *= av;
    }
    compiler_fence();
    // accumulate all 16 c-tiles
#pragma unroll
    for (int ct = 0; ct < 16; ++ct) {
      const bf16_t* vr = vb + (((size_t)b * CCH + ct * 16 + nl)) * NN + jj;
      v16bf vf = load_ab(vr + 16 * h, vr + 16 * h + 8);  // A[m=c][k=j]
      acc[ct] = wmma_bf16(vf, pf, acc[ct]);
    }
    compiler_fence();
  }
  // normalize by row sums (bounce L through LDS to column layout) and store
#pragma unroll
  for (int r = 0; r < 8; ++r) sw[r + 8 * h] = L[r];
  lds_fence();
  float rl = 1.0f / sw[nl];
#pragma unroll
  for (int ct = 0; ct < 16; ++ct)
#pragma unroll
    for (int r = 0; r < 8; ++r) {
      int c = ct * 16 + r + 8 * h;
      ob[(((size_t)b * CCH + c)) * NN + i0 + nl] = (bf16_t)(acc[ct][r] * rl);
    }
}

// ---------------------------------------------------------------------------
// K4: fused 1x1 conv over concat, computed without materializing the concat:
//     out[o,n] = sum_{k<256} WfL[o,k]*att_out[k,n]   (gamma pre-folded)
//              + sum_{k<256} WfR[o,k]*x[k,n] + bf[o]
//     Left half stages bf16 rows straight from ob (no conversion).
// ---------------------------------------------------------------------------
__global__ __launch_bounds__(256) void fuse_kernel(
    const float* __restrict__ x, const bf16_t* __restrict__ ob,
    const bf16_t* __restrict__ wfb, const float* __restrict__ bfp,
    float* __restrict__ out) {
  __shared__ alignas(16) bf16_t xs[8][32 * 16];
  int wid = threadIdx.x >> 5, lane = threadIdx.x & 31;
  int h = lane >> 4, nl = lane & 15;
  int wg = blockIdx.x * 8 + wid;
  int b = wg >> 12;
  int ot = (wg >> 8) & 15;
  int nt = wg & 255, n0 = nt << 4;
  bf16_t* xw = &xs[wid][0];

  v8f acc;
#pragma unroll
  for (int e = 0; e < 8; ++e) acc[e] = 0.f;

  for (int cb = 0; cb < 2 * CCH; cb += 32) {
    if (cb < CCH) {  // uniform branch: chunks never straddle the boundary
      // bf16 rows straight from attention output: 32 rows x 32B
#pragma unroll
      for (int t = 0; t < 2; ++t) {
        int slot = t * 32 + lane;
        int k = slot >> 1, hf = slot & 1;
        v8bf vv = *(const v8bf*)(ob + ((size_t)(b * CCH + cb + k)) * NN + n0 + hf * 8);
        *(v8bf*)(xw + k * 16 + hf * 8) = vv;
      }
    } else {
      // f32 rows from x, converted to bf16 during staging
#pragma unroll
      for (int t = 0; t < 4; ++t) {
        int slot = t * 32 + lane;
        int c = slot >> 2, q4 = slot & 3;
        float4 vv = *(const float4*)(x + ((size_t)(b * CCH + (cb - CCH) + c)) * NN + n0 + q4 * 4);
        v4bf bv;
        bv[0] = (bf16_t)vv.x; bv[1] = (bf16_t)vv.y;
        bv[2] = (bf16_t)vv.z; bv[3] = (bf16_t)vv.w;
        *(v4bf*)(xw + c * 16 + q4 * 4) = bv;
      }
    }
    lds_fence();
    v16bf xb;
#pragma unroll
    for (int idx = 0; idx < 16; ++idx) {
      int K = ((idx < 8) ? 0 : 16) + 8 * h + (idx & 7);
      xb[idx] = xw[K * 16 + nl];
    }
    compiler_fence();
    const bf16_t* wr = wfb + ((size_t)(ot * 16 + nl)) * (2 * CCH) + cb;
    v16bf wa = load_ab(wr + 8 * h, wr + 16 + 8 * h);
    acc = wmma_bf16(wa, xb, acc);
  }
#pragma unroll
  for (int r = 0; r < 8; ++r) {
    int o = ot * 16 + r + 8 * h;
    out[((size_t)b * CCH + o) * NN + n0 + nl] = acc[r] + bfp[o];
  }
}

// ---------------------------------------------------------------------------
extern "C" void kernel_launch(void* const* d_in, const int* in_sizes, int n_in,
                              void* d_out, int out_size, void* d_ws, size_t ws_size,
                              hipStream_t stream) {
  const float* x  = (const float*)d_in[0];
  const float* Wq = (const float*)d_in[1];
  const float* bq = (const float*)d_in[2];
  const float* Wk = (const float*)d_in[3];
  const float* bk = (const float*)d_in[4];
  const float* Wv = (const float*)d_in[5];
  const float* bv = (const float*)d_in[6];
  const float* gm = (const float*)d_in[7];
  const float* Wf = (const float*)d_in[8];
  const float* bf = (const float*)d_in[9];

  char* ws = (char*)d_ws;
  bf16_t* wqb = (bf16_t*)(ws + 0);          //  16 KB
  bf16_t* wkb = (bf16_t*)(ws + 16384);      //  16 KB
  bf16_t* wvb = (bf16_t*)(ws + 32768);      // 128 KB
  bf16_t* wfb = (bf16_t*)(ws + 163840);     // 256 KB (gamma folded in left half)
  bf16_t* qt  = (bf16_t*)(ws + 425984);     //   1 MB  [B][N][32]
  bf16_t* kt  = (bf16_t*)(ws + 1474560);    //   1 MB  [B][N][32]
  bf16_t* vbp = (bf16_t*)(ws + 2523136);    //   8 MB  [B][256][N]
  bf16_t* obp = (bf16_t*)(ws + 10911744);   //   8 MB  [B][256][N] bf16

  prep_kernel<<<512, 256, 0, stream>>>(Wq, Wk, Wv, Wf, gm, wqb, wkb, wvb, wfb);
  projqk_kernel<<<128, 256, 0, stream>>>(x, wqb, wkb, bq, bk, qt, kt);
  projv_kernel<<<2048, 256, 0, stream>>>(x, wvb, bv, vbp);
  attn_kernel<<<128, 256, 0, stream>>>(qt, kt, vbp, obp);
  fuse_kernel<<<2048, 256, 0, stream>>>(x, obp, wfb, bf, (float*)d_out);
}
